// Attention_66228395705099
// MI455X (gfx1250) — matrix operand order repaired and checked by reading, not compiled, not run on hardware
//
#include <hip/hip_runtime.h>

typedef __attribute__((ext_vector_type(16))) _Float16 v16h;
typedef __attribute__((ext_vector_type(8)))  _Float16 v8h;
typedef __attribute__((ext_vector_type(8)))  float    v8f;
typedef __attribute__((ext_vector_type(4)))  float    v4f;

#define S_LEN  2048
#define DMODEL 1024
#define NHEAD  16
#define HDIM   64
#define EPSV   1e-6f
#define LDT    40   // LDS tile pitch in halves: 32 data + 8 pad (20 dwords, conflict-free)

union AF { v16h v; v8h h[2]; };

// ---------------------------------------------------------------- f32 -> f16
__global__ void cvt_f32_f16(const float* __restrict__ src,
                            _Float16* __restrict__ dst, int n) {
  int i = blockIdx.x * blockDim.x + threadIdx.x;
  int stride = gridDim.x * blockDim.x;
  for (; i < n; i += stride) dst[i] = (_Float16)src[i];
}

// ------------------------------------------------------------- TDM tile fetch
#if __has_builtin(__builtin_amdgcn_tensor_load_to_lds) && \
    __has_builtin(__builtin_amdgcn_s_wait_tensorcnt)
#define USE_TDM 1
typedef __attribute__((ext_vector_type(4))) unsigned u32x4;
typedef __attribute__((ext_vector_type(8))) int      i32x8;
typedef __attribute__((ext_vector_type(4))) int      i32x4;

// Load a 128-row x 32-half tile (row stride = strideElems halves) from global
// into LDS at lds_addr, padding each 16-dword row with 4 dwords (pitch LDT=40).
__device__ __forceinline__ void tdm_load_tile(unsigned lds_addr,
                                              const _Float16* gsrc,
                                              unsigned strideElems) {
  const unsigned long long ga = (unsigned long long)(size_t)gsrc;
  u32x4 g0;
  g0[0] = 1u;                                   // count=1, user mode, no gather
  g0[1] = lds_addr;                             // LDS byte address
  g0[2] = (unsigned)(ga & 0xffffffffull);       // global_addr[31:0]
  g0[3] = (unsigned)((ga >> 32) & 0x01ffffffull) | (2u << 30); // [56:32] | type=2
  i32x8 g1;
  // data_size=1 (2B), pad_enable, pad_interval=3 (16 dw), pad_amount=3 (4 dw)
  g1[0] = (int)((1u << 16) | (1u << 20) | (3u << 22) | (3u << 25));
  const unsigned td0 = 1u << 20;                // tensor_dim0 (large, no OOB)
  const unsigned td1 = 1u << 20;                // tensor_dim1 (large, no OOB)
  g1[1] = (int)((td0 & 0xffffu) << 16);         // tensor_dim0[15:0]
  g1[2] = (int)(((td0 >> 16) & 0xffffu) | ((td1 & 0xffffu) << 16));
  g1[3] = (int)(((td1 >> 16) & 0xffffu) | (32u << 16));   // tile_dim0=32
  g1[4] = (int)128u;                            // tile_dim1=128, tile_dim2=0
  g1[5] = (int)strideElems;                     // tensor_dim0_stride[31:0]
  g1[6] = 0;                                    // stride hi, dim1_stride lo
  g1[7] = 0;
  i32x4 z4 = {0, 0, 0, 0};
  i32x8 z8 = {0, 0, 0, 0, 0, 0, 0, 0};
  // 6-arg form (clang-23 / therock-10.0 headers): g0, g1, g2, g3, extra, cpol
  __builtin_amdgcn_tensor_load_to_lds(g0, g1, z4, z4, z8, 0);
}
#else
#define USE_TDM 0
#endif

// ---------------------------------------------------------------- NT WMMA GEMM
// C[M,N] = A[M,K] * B[N,K]^T (both f16, K-contiguous), LDS double-buffered.
// Block tile 128x128: 8 waves, each 64x32 (4x2 of 16x16 WMMA tiles).
// MODE 0: f32 linear NT store + bias (out projection)
// MODE 1: f16 store, remap cols->heads [B,H,S,HD] + bias (Q,K proj)
// MODE 2: f16 store, remap to transposed heads [B,H,HD,S] + bias (V proj)
// MODE 3: f32 linear NT store, batched (scores), no bias, scaled
template <int MODE>
__global__ __launch_bounds__(256)
void gemm_nt_wmma(const _Float16* __restrict__ Aall,
                  const _Float16* __restrict__ Ball,
                  const float* __restrict__ bias,
                  void* __restrict__ Cout,
                  int M, int N, int K, int ldA, int ldB,
                  long sA, long sB, long sC, float scale) {
  __shared__ _Float16 smA[2][128 * LDT];
  __shared__ _Float16 smB[2][128 * LDT];

  const int z = blockIdx.z;
  const _Float16* Abase =
      Aall + (size_t)z * sA + (size_t)(blockIdx.x * 128) * ldA;
  const _Float16* Bbase =
      Ball + (size_t)z * sB + (size_t)(blockIdx.y * 128) * ldB;

  const int lane   = threadIdx.x & 31;
  const int wave   = threadIdx.x >> 5;
  const int laneLo = lane & 15;
  const int laneHi = lane >> 4;
  const int mW = (wave & 1) * 64;   // wave tile inside 128x128 block
  const int nW = (wave >> 1) * 32;

  v8f acc[4][2];
#pragma unroll
  for (int i = 0; i < 4; i++)
#pragma unroll
    for (int j = 0; j < 2; j++)
#pragma unroll
      for (int e = 0; e < 8; e++) acc[i][j][e] = 0.0f;

  const int KSTEPS = K >> 5;

#if USE_TDM
  if (wave == 0) {
    tdm_load_tile((unsigned)(size_t)&smA[0][0], Abase, (unsigned)ldA);
    tdm_load_tile((unsigned)(size_t)&smB[0][0], Bbase, (unsigned)ldB);
  }
#endif

  for (int ks = 0; ks < KSTEPS; ks++) {
#if USE_TDM
    const int cur = ks & 1;
    if (wave == 0) __builtin_amdgcn_s_wait_tensorcnt(0);
    __syncthreads(); // buf[cur] filled; buf[1-cur] free (all readers done)
    if (wave == 0 && ks + 1 < KSTEPS) {
      const int k0n = (ks + 1) << 5;
      tdm_load_tile((unsigned)(size_t)&smA[1 - cur][0], Abase + k0n,
                    (unsigned)ldA);
      tdm_load_tile((unsigned)(size_t)&smB[1 - cur][0], Bbase + k0n,
                    (unsigned)ldB);
    }
#else
    const int cur = 0;
    const int k0 = ks << 5;
    __syncthreads(); // previous iteration's readers done
    {
      // cooperative stage: 256 threads, each copies 32B of A and 32B of B
      const int row = threadIdx.x >> 1;
      const int seg = threadIdx.x & 1;
      const _Float16* sa = Abase + (size_t)row * ldA + k0 + seg * 16;
      *(v8h*)&smA[0][row * LDT + seg * 16]     = *(const v8h*)sa;
      *(v8h*)&smA[0][row * LDT + seg * 16 + 8] = *(const v8h*)(sa + 8);
      const _Float16* sb = Bbase + (size_t)row * ldB + k0 + seg * 16;
      *(v8h*)&smB[0][row * LDT + seg * 16]     = *(const v8h*)sb;
      *(v8h*)&smB[0][row * LDT + seg * 16 + 8] = *(const v8h*)(sb + 8);
    }
    __syncthreads();
#endif

    AF a[4], b[2];
#pragma unroll
    for (int mi = 0; mi < 4; mi++) {
      const _Float16* pa = &smA[cur][(mW + mi * 16 + laneLo) * LDT + laneHi * 8];
      a[mi].h[0] = *(const v8h*)pa;        // K = khalf*8 .. +7
      a[mi].h[1] = *(const v8h*)(pa + 16); // K = 16+khalf*8 .. +7
    }
#pragma unroll
    for (int ni = 0; ni < 2; ni++) {
      const _Float16* pb = &smB[cur][(nW + ni * 16 + laneLo) * LDT + laneHi * 8];
      b[ni].h[0] = *(const v8h*)pb;        // K = khalf*8 .. +7
      b[ni].h[1] = *(const v8h*)(pb + 16); // K = 16+khalf*8 .. +7
    }
#pragma unroll
    for (int mi = 0; mi < 4; mi++)
#pragma unroll
      for (int ni = 0; ni < 2; ni++)
        acc[mi][ni] = __builtin_amdgcn_wmma_f32_16x16x32_f16(
            false, a[mi].v, false, b[ni].v, (short)0, acc[mi][ni], false, false);

#if USE_TDM
    __syncthreads(); // all waves done reading buf[cur] before it is refilled
#endif
  }

#pragma unroll
  for (int mi = 0; mi < 4; mi++) {
#pragma unroll
    for (int ni = 0; ni < 2; ni++) {
#pragma unroll
      for (int r = 0; r < 8; r++) {
        const int gm = blockIdx.x * 128 + mW + mi * 16 + laneHi * 8 + r;
        const int gn = blockIdx.y * 128 + nW + ni * 16 + laneLo;
        float val = acc[mi][ni][r] * scale;
        if (MODE == 0) {
          __builtin_nontemporal_store(
              val + bias[gn],
              (float*)Cout + (size_t)z * sC + (size_t)gm * N + gn);
        } else if (MODE == 3) {
          __builtin_nontemporal_store(
              val, (float*)Cout + (size_t)z * sC + (size_t)gm * N + gn);
        } else if (MODE == 1) {
          const int bb = gm >> 11, ss = gm & (S_LEN - 1);
          const int hh = gn >> 6,  ee = gn & (HDIM - 1);
          const size_t idx =
              ((((size_t)bb * NHEAD + hh) * S_LEN) + ss) * HDIM + ee;
          ((_Float16*)Cout)[idx] = (_Float16)(val + bias[gn]);
        } else { // MODE 2: transposed-V heads
          const int bb = gm >> 11, ss = gm & (S_LEN - 1);
          const int hh = gn >> 6,  ee = gn & (HDIM - 1);
          const size_t idx =
              ((((size_t)bb * NHEAD + hh) * HDIM) + ee) * (size_t)S_LEN + ss;
          ((_Float16*)Cout)[idx] = (_Float16)(val + bias[gn]);
        }
      }
    }
  }
}

// ------------------------------------------------- softmax + diag mask + renorm
__global__ __launch_bounds__(256)
void softmax_mask_renorm(float* __restrict__ s) {
  const size_t row = blockIdx.x;
  float* p = s + row * (size_t)S_LEN;
  const int diag = (int)(row & (S_LEN - 1));
  const int tid = threadIdx.x;
  __shared__ float red[256];

  float vals[8];
  float mx = -3.4e38f;
#pragma unroll
  for (int i = 0; i < 8; i++) {
    vals[i] = __builtin_nontemporal_load(p + i * 256 + tid);
    mx = fmaxf(mx, vals[i]);
  }
  red[tid] = mx;
  __syncthreads();
  for (int off = 128; off > 0; off >>= 1) {
    if (tid < off) red[tid] = fmaxf(red[tid], red[tid + off]);
    __syncthreads();
  }
  mx = red[0];
  __syncthreads();

  float sum = 0.f;
#pragma unroll
  for (int i = 0; i < 8; i++) {
    vals[i] = __expf(vals[i] - mx);
    sum += vals[i];
  }
  red[tid] = sum;
  __syncthreads();
  for (int off = 128; off > 0; off >>= 1) {
    if (tid < off) red[tid] += red[tid + off];
    __syncthreads();
  }
  const float T = red[0];

  // reference: p = softmax; p[diag]=0; p /= (sum(p) + EPS)
  const float ed  = __expf(p[diag] - mx);
  const float r   = (T - ed) / T;
  const float inv = 1.0f / (T * (r + EPSV));
#pragma unroll
  for (int i = 0; i < 8; i++) {
    const int j = i * 256 + tid;
    __builtin_nontemporal_store((j == diag) ? 0.0f : vals[i] * inv, p + j);
  }
}

// ---------------------------------------------------------------- attn = s @ V
// A = s (f32, NT-loaded, converted to f16 in regs), B = Vt (f16, [HD,S])
// Block tile 128x64: 8 waves as 4(M)x2(N), each wave 32x32 (2x2 WMMA tiles).
__global__ __launch_bounds__(256)
void attn_v_wmma(const float* __restrict__ Sall,
                 const _Float16* __restrict__ Vt,
                 _Float16* __restrict__ attn) {
  const int z  = blockIdx.z; // b*NHEAD + h
  const int bb = z >> 4, hh = z & 15;
  const float*    Sp = Sall + (size_t)z * S_LEN * S_LEN;
  const _Float16* Bm = Vt   + (size_t)z * HDIM * S_LEN;

  const int lane   = threadIdx.x & 31;
  const int wave   = threadIdx.x >> 5;
  const int laneLo = lane & 15;
  const int laneHi = lane >> 4;
  const int m0 = blockIdx.x * 128 + (wave >> 1) * 32;
  const int n0 = (wave & 1) * 32;

  v8f acc[2][2];
#pragma unroll
  for (int i = 0; i < 2; i++)
#pragma unroll
    for (int j = 0; j < 2; j++)
#pragma unroll
      for (int e = 0; e < 8; e++) acc[i][j][e] = 0.0f;

  for (int k0 = 0; k0 < S_LEN; k0 += 32) {
    AF a[2], b[2];
#pragma unroll
    for (int mi = 0; mi < 2; mi++) {
      const float* pa =
          Sp + (size_t)(m0 + mi * 16 + laneLo) * S_LEN + k0 + laneHi * 8;
      const v4f f0 = __builtin_nontemporal_load((const v4f*)pa);
      const v4f f1 = __builtin_nontemporal_load((const v4f*)(pa + 4));
      const v4f f2 = __builtin_nontemporal_load((const v4f*)(pa + 16));
      const v4f f3 = __builtin_nontemporal_load((const v4f*)(pa + 20));
#pragma unroll
      for (int e = 0; e < 4; e++) {
        a[mi].v[e]      = (_Float16)f0[e];
        a[mi].v[4 + e]  = (_Float16)f1[e];
        a[mi].v[8 + e]  = (_Float16)f2[e];
        a[mi].v[12 + e] = (_Float16)f3[e];
      }
    }
#pragma unroll
    for (int ni = 0; ni < 2; ni++) {
      const _Float16* pb =
          Bm + (size_t)(n0 + ni * 16 + laneLo) * S_LEN + k0 + laneHi * 8;
      b[ni].h[0] = *(const v8h*)pb;
      b[ni].h[1] = *(const v8h*)(pb + 16);
    }
#pragma unroll
    for (int mi = 0; mi < 2; mi++)
#pragma unroll
      for (int ni = 0; ni < 2; ni++)
        acc[mi][ni] = __builtin_amdgcn_wmma_f32_16x16x32_f16(
            false, a[mi].v, false, b[ni].v, (short)0, acc[mi][ni], false, false);
  }

#pragma unroll
  for (int mi = 0; mi < 2; mi++)
#pragma unroll
    for (int ni = 0; ni < 2; ni++)
#pragma unroll
      for (int r = 0; r < 8; r++) {
        const int gm = m0 + mi * 16 + laneHi * 8 + r;
        const int gn = n0 + ni * 16 + laneLo;
        attn[((size_t)bb * S_LEN + gm) * DMODEL + hh * HDIM + gn] =
            (_Float16)acc[mi][ni][r];
      }
}

// ----------------------------------------------------------------------------
extern "C" void kernel_launch(void* const* d_in, const int* in_sizes, int n_in,
                              void* d_out, int out_size, void* d_ws,
                              size_t ws_size, hipStream_t stream) {
  const float* q  = (const float*)d_in[0];
  const float* k  = (const float*)d_in[1];
  const float* v  = (const float*)d_in[2];
  const float* Wq = (const float*)d_in[3];
  const float* bq = (const float*)d_in[4];
  const float* Wk = (const float*)d_in[5];
  const float* bk = (const float*)d_in[6];
  const float* Wv = (const float*)d_in[7];
  const float* bv = (const float*)d_in[8];
  const float* Wo = (const float*)d_in[9];
  const float* bo = (const float*)d_in[10];

  float* out  = (float*)d_out;
  float* sreg = out + (size_t)2 * S_LEN * DMODEL; // tuple output #2

  const size_t NQ = (size_t)2 * S_LEN * DMODEL; // 4,194,304
  const size_t NW = (size_t)DMODEL * DMODEL;    // 1,048,576
  _Float16* w    = (_Float16*)d_ws;
  _Float16* q16  = w;
  _Float16* k16  = w + NQ;
  _Float16* v16  = w + 2 * NQ;
  _Float16* Wq16 = w + 3 * NQ;
  _Float16* Wk16 = Wq16 + NW;
  _Float16* Wv16 = Wk16 + NW;
  _Float16* Wo16 = Wv16 + NW;
  _Float16* Qh   = Wo16 + NW; // [B,H,S,HD]
  _Float16* Kh   = Qh + NQ;   // [B,H,S,HD]
  _Float16* Vt   = Kh + NQ;   // [B,H,HD,S]
  _Float16* at16 = Vt + NQ;   // [B,S,D]

  // 1) f32 -> f16 conversions
  cvt_f32_f16<<<2048, 256, 0, stream>>>(q, q16, (int)NQ);
  cvt_f32_f16<<<2048, 256, 0, stream>>>(k, k16, (int)NQ);
  cvt_f32_f16<<<2048, 256, 0, stream>>>(v, v16, (int)NQ);
  cvt_f32_f16<<<1024, 256, 0, stream>>>(Wq, Wq16, (int)NW);
  cvt_f32_f16<<<1024, 256, 0, stream>>>(Wk, Wk16, (int)NW);
  cvt_f32_f16<<<1024, 256, 0, stream>>>(Wv, Wv16, (int)NW);
  cvt_f32_f16<<<1024, 256, 0, stream>>>(Wo, Wo16, (int)NW);

  dim3 blk(256);

  // 2) QKV projections: M=4096, N=1024, K=1024 (NT, bias fused)
  dim3 gProj(4096 / 128, 1024 / 128, 1);
  gemm_nt_wmma<1><<<gProj, blk, 0, stream>>>(q16, Wq16, bq, Qh, 4096, 1024,
                                             1024, 1024, 1024, 0, 0, 0, 1.0f);
  gemm_nt_wmma<1><<<gProj, blk, 0, stream>>>(k16, Wk16, bk, Kh, 4096, 1024,
                                             1024, 1024, 1024, 0, 0, 0, 1.0f);
  gemm_nt_wmma<2><<<gProj, blk, 0, stream>>>(v16, Wv16, bv, Vt, 4096, 1024,
                                             1024, 1024, 1024, 0, 0, 0, 1.0f);

  // 3) scores = Q Kᵀ / 8, batched over B*H, f32 NT into s output region
  dim3 gScore(2048 / 128, 2048 / 128, 32);
  gemm_nt_wmma<3><<<gScore, blk, 0, stream>>>(
      Qh, Kh, nullptr, sreg, 2048, 2048, 64, 64, 64,
      (long)S_LEN * HDIM, (long)S_LEN * HDIM, (long)S_LEN * S_LEN, 0.125f);

  // 4) softmax + diag-mask + renormalize, in place on s
  softmax_mask_renorm<<<2 * NHEAD * S_LEN, 256, 0, stream>>>(sreg);

  // 5) attn = s @ V (per head), merge heads to [B,S,D] f16
  dim3 gAttn(2048 / 128, 1, 32);
  attn_v_wmma<<<gAttn, blk, 0, stream>>>(sreg, Vt, at16);

  // 6) out = attn @ Woᵀ + bo, f32
  dim3 gOut(4096 / 128, 1024 / 128, 1);
  gemm_nt_wmma<0><<<gOut, blk, 0, stream>>>(at16, Wo16, bo, out, 4096, 1024,
                                            1024, 1024, 1024, 0, 0, 0, 1.0f);
}
